// Block_83880711291003
// MI455X (gfx1250) — matrix operand order repaired and checked by reading, not compiled, not run on hardware
//
#include <hip/hip_runtime.h>
#include <hip/hip_bf16.h>
#include <math.h>

typedef __bf16 bf16;
typedef __attribute__((ext_vector_type(16))) __bf16 v16bf;
typedef __attribute__((ext_vector_type(8)))  __bf16 v8bf;
typedef __attribute__((ext_vector_type(8)))  float  v8f;

#define D_MODEL 512
#define T_SEQ   2048
#define BATCH   4
#define NHEAD   8
#define DKH     64
#define ROWS    (BATCH * T_SEQ)          // 8192
#define D_FF    (4 * D_MODEL)            // 2048

// ---------------------------------------------------------------- WMMA core
static __device__ __forceinline__ v8f wmma_bf16f32(v16bf a, v16bf b, v8f c) {
  // v_wmma_f32_16x16x32_bf16: D = A(16x32) * B(32x16) + C(16x16 f32)
  return __builtin_amdgcn_wmma_f32_16x16x32_bf16(
      /*neg_a=*/false, a, /*neg_b=*/false, b,
      /*c_mod=*/(short)0, c, /*reuse_a=*/false, /*reuse_b=*/false);
}

// A fragment: rows m0..m0+15, K slab k0..k0+31 of a row-major [.., ld] matrix.
// ISA layout: lane L<16 -> M=L, elems 0..7 = K 0..7, elems 8..15 = K 16..23;
//             lane L>=16 -> M=L-16, K offsets +8.
static __device__ __forceinline__ v16bf load_a_frag(const bf16* base, int ld) {
  int lane = threadIdx.x & 31;
  int mrow = lane & 15;
  int kb   = (lane >> 4) << 3;                     // 0 or 8
  const bf16* p = base + (size_t)mrow * ld + kb;
  v8bf lo = *(const v8bf*)(p);
  v8bf hi = *(const v8bf*)(p + 16);
  v16bf a;
#pragma unroll
  for (int i = 0; i < 8; ++i) { a[i] = lo[i]; a[i + 8] = hi[i]; }
  return a;
}

// B fragment: B is K(32) x N(16); source stored as Bt[N][K] row-major (ld=K).
// ISA layout: lane&15 = N column, elems 0..7 = K 8h..8h+7, elems 8..15 = K 16+8h..23+8h (h = lane>=16).
static __device__ __forceinline__ v16bf load_b_frag(const bf16* base, int ld) {
  int lane = threadIdx.x & 31;
  int n  = lane & 15;
  int kb = (lane >> 4) << 3;                       // 0 or 8
  const bf16* p = base + (size_t)n * ld + kb;
  v8bf lo = *(const v8bf*)(p);
  v8bf hi = *(const v8bf*)(p + 16);
  v16bf b;
#pragma unroll
  for (int i = 0; i < 8; ++i) { b[i] = lo[i]; b[i + 8] = hi[i]; }
  return b;
}

// ------------------------------------------------------------- LayerNorm
// one wave per 512-wide row; each lane owns 16 contiguous elements.
__global__ __launch_bounds__(256) void ln_kernel(
    const float* __restrict__ x, const float* __restrict__ g,
    const float* __restrict__ be, bf16* __restrict__ out, int rows) {
  int wave = (int)((blockIdx.x * blockDim.x + threadIdx.x) >> 5);
  if (wave >= rows) return;
  int lane = threadIdx.x & 31;
  const float* xr = x + (size_t)wave * D_MODEL;
  float v[16];
  float s = 0.f;
#pragma unroll
  for (int i = 0; i < 16; ++i) { v[i] = xr[lane * 16 + i]; s += v[i]; }
#pragma unroll
  for (int m = 1; m < 32; m <<= 1) s += __shfl_xor(s, m, 32);
  float mu = s * (1.0f / D_MODEL);
  float vs = 0.f;
#pragma unroll
  for (int i = 0; i < 16; ++i) { float d = v[i] - mu; vs += d * d; }
#pragma unroll
  for (int m = 1; m < 32; m <<= 1) vs += __shfl_xor(vs, m, 32);
  float rstd = rsqrtf(vs * (1.0f / D_MODEL) + 1e-5f);
  bf16* orow = out + (size_t)wave * D_MODEL;
#pragma unroll
  for (int i = 0; i < 16; ++i) {
    int c = lane * 16 + i;
    orow[c] = (bf16)((v[i] - mu) * rstd * g[c] + be[c]);
  }
}

// ----------------------------------------------- weight convert / transpose
// Wq/Wk/Wv [H,512,64] f32 -> Wt[h*64+kk][d] bf16 ([512,512], N-major rows)
__global__ __launch_bounds__(256) void cvt_w_qkv(const float* __restrict__ W,
                                                 bf16* __restrict__ Wt) {
  int idx = blockIdx.x * blockDim.x + threadIdx.x;
  if (idx >= D_MODEL * D_MODEL) return;
  int d = idx & (D_MODEL - 1);
  int n = idx >> 9;                 // h*64+kk
  int h = n >> 6, kk = n & 63;
  Wt[idx] = (bf16)W[((size_t)h * D_MODEL + d) * DKH + kk];
}

// W [K,N] f32 -> Wt [N,K] bf16
__global__ __launch_bounds__(256) void cvt_w_t(const float* __restrict__ W,
                                               bf16* __restrict__ Wt,
                                               int K, int N) {
  int idx = blockIdx.x * blockDim.x + threadIdx.x;
  if (idx >= N * K) return;
  int k = idx % K, n = idx / K;
  Wt[idx] = (bf16)W[(size_t)k * N + n];
}

// v [B,T,H*64] bf16 -> vt [B,H,64,T] bf16 (so PV B-fragments are contiguous)
__global__ __launch_bounds__(256) void transpose_v_kernel(
    const bf16* __restrict__ v, bf16* __restrict__ vt) {
  int idx = blockIdx.x * blockDim.x + threadIdx.x;
  if (idx >= BATCH * NHEAD * DKH * T_SEQ) return;
  int t = idx & (T_SEQ - 1);
  int r = idx >> 11;
  int kk = r & 63; r >>= 6;
  int h = r & 7;  int b = r >> 3;
  vt[idx] = v[((size_t)(b * T_SEQ + t)) * D_MODEL + h * DKH + kk];
}

// ------------------------------------------------------------------- GEMM
// A bf16 [M,K] row-major, Bt bf16 [N,K] row-major, out = A*B (+bias)(+resid)
// flags: bit0 store bf16 (else f32), bit1 relu
// wave tile: 32(M) x 64(N), register double-buffered K loop (K%32==0, M%32==0)
__global__ __launch_bounds__(128) void gemm_bf16_kernel(
    const bf16* __restrict__ A, const bf16* __restrict__ Bt,
    const float* __restrict__ bias, const float* __restrict__ resid,
    void* __restrict__ out, int M, int N, int K, int flags) {
  int wave = (int)((blockIdx.x * blockDim.x + threadIdx.x) >> 5);
  int nt = N >> 6;
  int mt = wave / nt, ntile = wave - mt * nt;
  if (mt >= (M >> 5)) return;
  int m0 = mt << 5, n0 = ntile << 6;

  v8f acc[2][4];
#pragma unroll
  for (int s = 0; s < 2; ++s)
#pragma unroll
    for (int t = 0; t < 4; ++t) acc[s][t] = (v8f)0.f;

  const bf16* arow0 = A + (size_t)m0 * K;
  const bf16* arow1 = A + (size_t)(m0 + 16) * K;
  const bf16* brow0 = Bt + (size_t)(n0 +  0) * K;
  const bf16* brow1 = Bt + (size_t)(n0 + 16) * K;
  const bf16* brow2 = Bt + (size_t)(n0 + 32) * K;
  const bf16* brow3 = Bt + (size_t)(n0 + 48) * K;

  // prologue: fragments for k0 = 0
  v16bf a0 = load_a_frag(arow0, K);
  v16bf a1 = load_a_frag(arow1, K);
  v16bf b0 = load_b_frag(brow0, K);
  v16bf b1 = load_b_frag(brow1, K);
  v16bf b2 = load_b_frag(brow2, K);
  v16bf b3 = load_b_frag(brow3, K);

  for (int k0 = 0; k0 < K; k0 += 32) {
    // prefetch next K slab (last iteration redundantly reloads current slab)
    int kn = (k0 + 32 < K) ? (k0 + 32) : k0;
    v16bf na0 = load_a_frag(arow0 + kn, K);
    v16bf na1 = load_a_frag(arow1 + kn, K);
    v16bf nb0 = load_b_frag(brow0 + kn, K);
    v16bf nb1 = load_b_frag(brow1 + kn, K);
    v16bf nb2 = load_b_frag(brow2 + kn, K);
    v16bf nb3 = load_b_frag(brow3 + kn, K);

    acc[0][0] = wmma_bf16f32(a0, b0, acc[0][0]);
    acc[0][1] = wmma_bf16f32(a0, b1, acc[0][1]);
    acc[0][2] = wmma_bf16f32(a0, b2, acc[0][2]);
    acc[0][3] = wmma_bf16f32(a0, b3, acc[0][3]);
    acc[1][0] = wmma_bf16f32(a1, b0, acc[1][0]);
    acc[1][1] = wmma_bf16f32(a1, b1, acc[1][1]);
    acc[1][2] = wmma_bf16f32(a1, b2, acc[1][2]);
    acc[1][3] = wmma_bf16f32(a1, b3, acc[1][3]);

    a0 = na0; a1 = na1;
    b0 = nb0; b1 = nb1; b2 = nb2; b3 = nb3;
  }

  int lane = threadIdx.x & 31;
  int nl = lane & 15, hif = lane >> 4;
#pragma unroll
  for (int s = 0; s < 2; ++s) {
#pragma unroll
    for (int t = 0; t < 4; ++t) {
      int n = n0 + 16 * t + nl;
      float bv = bias ? bias[n] : 0.f;
#pragma unroll
      for (int r = 0; r < 8; ++r) {
        int row = m0 + 16 * s + r + hif * 8;
        float val = acc[s][t][r] + bv;
        if (flags & 2) val = fmaxf(val, 0.f);
        if (resid) val += resid[(size_t)row * N + n];
        if (flags & 1) ((bf16*)out)[(size_t)row * N + n] = (bf16)val;
        else           ((float*)out)[(size_t)row * N + n] = val;
      }
    }
  }
}

// ------------------------------------------------------------ Flash attention
// one wave per 16-row Q tile of one (b,h); K/V streamed (L2-resident).
__global__ __launch_bounds__(128) void flash_attn_kernel(
    const bf16* __restrict__ q, const bf16* __restrict__ k,
    const bf16* __restrict__ vt, bf16* __restrict__ o) {
  __shared__ __align__(16) bf16 pshare[4][16 * 32];   // P staging, 1KB/wave
  int wid = threadIdx.x >> 5;
  int wave = blockIdx.x * 4 + wid;
  const int QT = T_SEQ / 16;                          // 128
  int qt = wave % QT;
  int bh = wave / QT;
  int h = bh & 7, b = bh >> 3;
  int i0 = qt * 16;
  int lane = threadIdx.x & 31;
  int nl = lane & 15, hif = lane >> 4;

  // Q fragments, pre-scaled by 1/sqrt(d_k)=0.125 (mask value already scaled)
  const bf16* qbase = q + ((size_t)(b * T_SEQ + i0)) * D_MODEL + h * DKH;
  v16bf aq0 = load_a_frag(qbase, D_MODEL);
  v16bf aq1 = load_a_frag(qbase + 32, D_MODEL);
#pragma unroll
  for (int i = 0; i < 16; ++i) {
    aq0[i] = (bf16)((float)aq0[i] * 0.125f);
    aq1[i] = (bf16)((float)aq1[i] * 0.125f);
  }

  float mrow[8], lrow[8];
  v8f oacc[4];
#pragma unroll
  for (int r = 0; r < 8; ++r) { mrow[r] = -3.0e38f; lrow[r] = 0.f; }
#pragma unroll
  for (int t = 0; t < 4; ++t) oacc[t] = (v8f)0.f;

  const bf16* kbase = k + ((size_t)b * T_SEQ) * D_MODEL + h * DKH;
  const bf16* vbase = vt + ((size_t)(b * NHEAD + h)) * DKH * T_SEQ;

  for (int j0 = 0; j0 <= i0 + 15; j0 += 32) {
    // S = Q * K^T  (two 16x16 tiles, d_k split into two K=32 slabs)
    v8f s0 = (v8f)0.f, s1 = (v8f)0.f;
    s0 = wmma_bf16f32(aq0, load_b_frag(kbase + (size_t)j0 * D_MODEL, D_MODEL), s0);
    s0 = wmma_bf16f32(aq1, load_b_frag(kbase + (size_t)j0 * D_MODEL + 32, D_MODEL), s0);
    s1 = wmma_bf16f32(aq0, load_b_frag(kbase + (size_t)(j0 + 16) * D_MODEL, D_MODEL), s1);
    s1 = wmma_bf16f32(aq1, load_b_frag(kbase + (size_t)(j0 + 16) * D_MODEL + 32, D_MODEL), s1);

    // hoist V fragments: independent of softmax, overlap the exp/shuffle work
    v16bf bv0 = load_b_frag(vbase + (size_t)( 0) * T_SEQ + j0, T_SEQ);
    v16bf bv1 = load_b_frag(vbase + (size_t)(16) * T_SEQ + j0, T_SEQ);
    v16bf bv2 = load_b_frag(vbase + (size_t)(32) * T_SEQ + j0, T_SEQ);
    v16bf bv3 = load_b_frag(vbase + (size_t)(48) * T_SEQ + j0, T_SEQ);

    // causal mask (reference: -1e9 applied before /sqrt(dk) -> -1.25e8)
    if (j0 + 31 > i0) {
#pragma unroll
      for (int r = 0; r < 8; ++r) {
        int row = i0 + r + hif * 8;
        if (j0 + nl > row)      s0[r] = -1.25e8f;
        if (j0 + 16 + nl > row) s1[r] = -1.25e8f;
      }
    }

    // online softmax over this 32-col block (row stats per 16-lane half)
    float pr0[8], pr1[8];
#pragma unroll
    for (int r = 0; r < 8; ++r) {
      float mb = fmaxf(s0[r], s1[r]);
#pragma unroll
      for (int m = 1; m < 16; m <<= 1) mb = fmaxf(mb, __shfl_xor(mb, m, 32));
      float mn = fmaxf(mrow[r], mb);
      float scale = __expf(mrow[r] - mn);
      mrow[r] = mn;
      float p0 = __expf(s0[r] - mn);
      float p1 = __expf(s1[r] - mn);
      float ps = p0 + p1;
#pragma unroll
      for (int m = 1; m < 16; m <<= 1) ps += __shfl_xor(ps, m, 32);
      lrow[r] = lrow[r] * scale + ps;
      pr0[r] = p0; pr1[r] = p1;
#pragma unroll
      for (int t = 0; t < 4; ++t) oacc[t][r] *= scale;
    }

    // C-layout -> A-layout through LDS (wave-private 16x32 tile)
    bf16* pl = pshare[wid];
#pragma unroll
    for (int r = 0; r < 8; ++r) {
      int row = r + hif * 8;
      pl[row * 32 + nl]      = (bf16)pr0[r];
      pl[row * 32 + 16 + nl] = (bf16)pr1[r];
    }
    asm volatile("s_wait_dscnt 0x0" ::: "memory");   // same-wave LDS RAW
    v16bf ap = load_a_frag(pl, 32);

    // O += P * V   (V pre-transposed: Bt[d][t], ld = T)
    oacc[0] = wmma_bf16f32(ap, bv0, oacc[0]);
    oacc[1] = wmma_bf16f32(ap, bv1, oacc[1]);
    oacc[2] = wmma_bf16f32(ap, bv2, oacc[2]);
    oacc[3] = wmma_bf16f32(ap, bv3, oacc[3]);
  }

  // normalize and store O (bf16, [B,T,H*64] layout)
  bf16* obase = o + ((size_t)(b * T_SEQ + i0)) * D_MODEL + h * DKH;
#pragma unroll
  for (int t = 0; t < 4; ++t) {
#pragma unroll
    for (int r = 0; r < 8; ++r) {
      int row = r + hif * 8;
      float val = oacc[t][r] / lrow[r];
      obase[(size_t)row * D_MODEL + 16 * t + nl] = (bf16)val;
    }
  }
}

// ------------------------------------------------------------------ launch
extern "C" void kernel_launch(void* const* d_in, const int* in_sizes, int n_in,
                              void* d_out, int out_size, void* d_ws, size_t ws_size,
                              hipStream_t stream) {
  const float* x   = (const float*)d_in[0];
  const float* Wq  = (const float*)d_in[1];
  const float* bq  = (const float*)d_in[2];
  const float* Wk  = (const float*)d_in[3];
  const float* bk  = (const float*)d_in[4];
  const float* Wv  = (const float*)d_in[5];
  const float* bv  = (const float*)d_in[6];
  const float* Wp  = (const float*)d_in[7];
  const float* bp  = (const float*)d_in[8];
  const float* W1  = (const float*)d_in[9];
  const float* b1  = (const float*)d_in[10];
  const float* W2  = (const float*)d_in[11];
  const float* b2  = (const float*)d_in[12];
  const float* g1  = (const float*)d_in[13];
  const float* be1 = (const float*)d_in[14];
  const float* g2  = (const float*)d_in[15];
  const float* be2 = (const float*)d_in[16];
  float* out = (float*)d_out;

  const size_t MB = 1024ull * 1024ull;
  char* ws = (char*)d_ws;
  bf16* hln = (bf16*)(ws + 0 * MB);        // [8192,512]  (reused as h2)
  bf16* qb  = (bf16*)(ws + 8 * MB);        // [8192,512]
  bf16* kb  = (bf16*)(ws + 16 * MB);       // [8192,512]
  bf16* vb  = (bf16*)(ws + 24 * MB);       // [8192,512]
  bf16* vtb = (bf16*)(ws + 32 * MB);       // [B,H,64,T]
  bf16* ob  = (bf16*)(ws + 40 * MB);       // [8192,512]
  float* x2 = (float*)(ws + 48 * MB);      // [8192,512] f32
  bf16* ff1 = (bf16*)(ws + 8 * MB);        // [8192,2048] reuses q/k/v/vt
  bf16* WqT = (bf16*)(ws + 64 * MB);
  bf16* WkT = (bf16*)(ws + 64 * MB + 512 * 1024);
  bf16* WvT = (bf16*)(ws + 65 * MB);
  bf16* WpT = (bf16*)(ws + 65 * MB + 512 * 1024);
  bf16* W1T = (bf16*)(ws + 66 * MB);       // [2048,512]
  bf16* W2T = (bf16*)(ws + 68 * MB);       // [512,2048]

  // weight conversion / transpose (bf16, [N,K] layouts)
  cvt_w_qkv<<<(D_MODEL * D_MODEL + 255) / 256, 256, 0, stream>>>(Wq, WqT);
  cvt_w_qkv<<<(D_MODEL * D_MODEL + 255) / 256, 256, 0, stream>>>(Wk, WkT);
  cvt_w_qkv<<<(D_MODEL * D_MODEL + 255) / 256, 256, 0, stream>>>(Wv, WvT);
  cvt_w_t<<<(D_MODEL * D_MODEL + 255) / 256, 256, 0, stream>>>(Wp, WpT, D_MODEL, D_MODEL);
  cvt_w_t<<<(D_MODEL * D_FF + 255) / 256, 256, 0, stream>>>(W1, W1T, D_MODEL, D_FF);
  cvt_w_t<<<(D_FF * D_MODEL + 255) / 256, 256, 0, stream>>>(W2, W2T, D_FF, D_MODEL);

  // 1) h = LN1(x)
  ln_kernel<<<ROWS / 8, 256, 0, stream>>>(x, g1, be1, hln, ROWS);

  // 2-4) q/k/v = h @ W*T + b*   (M=8192, N=512, K=512; 32x64 wave tiles)
  {
    int waves = (ROWS / 32) * (D_MODEL / 64);
    gemm_bf16_kernel<<<waves / 4, 128, 0, stream>>>(hln, WqT, bq, nullptr, qb,
                                                    ROWS, D_MODEL, D_MODEL, 1);
    gemm_bf16_kernel<<<waves / 4, 128, 0, stream>>>(hln, WkT, bk, nullptr, kb,
                                                    ROWS, D_MODEL, D_MODEL, 1);
    gemm_bf16_kernel<<<waves / 4, 128, 0, stream>>>(hln, WvT, bv, nullptr, vb,
                                                    ROWS, D_MODEL, D_MODEL, 1);
  }

  // 5) vt = head-transpose(v)
  transpose_v_kernel<<<(BATCH * NHEAD * DKH * T_SEQ + 255) / 256, 256, 0, stream>>>(vb, vtb);

  // 6) o = causal-softmax(qk^T) v  (flash, 1 wave / 16-row tile)
  {
    int waves = BATCH * NHEAD * (T_SEQ / 16);   // 4096
    flash_attn_kernel<<<waves / 4, 128, 0, stream>>>(qb, kb, vtb, ob);
  }

  // 7) x2 = x + o @ Wp + bp  (f32 out)
  {
    int waves = (ROWS / 32) * (D_MODEL / 64);
    gemm_bf16_kernel<<<waves / 4, 128, 0, stream>>>(ob, WpT, bp, x, x2,
                                                    ROWS, D_MODEL, D_MODEL, 0);
  }

  // 8) h2 = LN2(x2)
  ln_kernel<<<ROWS / 8, 256, 0, stream>>>(x2, g2, be2, hln, ROWS);

  // 9) ff1 = relu(h2 @ W1 + b1)  (M=8192, N=2048, K=512)
  {
    int waves = (ROWS / 32) * (D_FF / 64);
    gemm_bf16_kernel<<<waves / 4, 128, 0, stream>>>(hln, W1T, b1, nullptr, ff1,
                                                    ROWS, D_FF, D_MODEL, 1 | 2);
  }

  // 10) out = x2 + ff1 @ W2 + b2  (M=8192, N=512, K=2048, f32 out)
  {
    int waves = (ROWS / 32) * (D_MODEL / 64);
    gemm_bf16_kernel<<<waves / 4, 128, 0, stream>>>(ff1, W2T, b2, x2, out,
                                                    ROWS, D_MODEL, D_FF, 0);
  }
}
